// ExpertModel_4303557231421
// MI455X (gfx1250) — matrix operand order repaired and checked by reading, not compiled, not run on hardware
//
#include <hip/hip_runtime.h>

// ---------------------------------------------------------------------------
// MI455X (gfx1250, wave32) persistent-rollout kernel.
// Each wave32 owns 16 trajectories and runs all 256 Euler steps.
// 64x64 layers (fwd + input-grad VJP) run on v_wmma_f32_16x16x32_f16.
// Transcendentals use v_exp_f32 + v_rcp_f32 (no IEEE divide chains).
// ---------------------------------------------------------------------------

typedef __attribute__((ext_vector_type(16))) _Float16 v16h;
typedef __attribute__((ext_vector_type(8)))  float    v8f;

#define B_TOT 16384
#define T_STEPS 256
#define HD 64
#define WAVES_PER_BLK 4
#define ROWS_PER_WAVE 16
#define TSTR 72            // LDS tile row stride (64 + 8 pad)

__device__ __forceinline__ float fast_rcp(float x) {
    return __builtin_amdgcn_rcpf(x);       // v_rcp_f32 (~1 ulp)
}
__device__ __forceinline__ float fast_sig(float z) {
    return fast_rcp(1.0f + __expf(-z));
}
__device__ __forceinline__ float fast_tanh(float z) {
    return 1.0f - 2.0f * fast_rcp(__expf(2.0f * z) + 1.0f);
}

__global__ __launch_bounds__(128)
void motor_rollout_kernel(const float* __restrict__ init_obs,
                          const float* __restrict__ actions,
                          const float* __restrict__ Wd1, const float* __restrict__ bd1,
                          const float* __restrict__ Wd2, const float* __restrict__ bd2,
                          const float* __restrict__ Wd3, const float* __restrict__ bd3,
                          const float* __restrict__ Wq1, const float* __restrict__ bq1,
                          const float* __restrict__ Wq2, const float* __restrict__ bq2,
                          const float* __restrict__ Wq3, const float* __restrict__ bq3,
                          float* __restrict__ out)
{
    constexpr float I_LIM     = 250.0f;
    constexpr float INV_I_LIM = 1.0f / 250.0f;
    constexpr float U_MAX     = 230.94010767585033f;   // 400/sqrt(3)
    constexpr float R_S       = 0.05f;
    constexpr float OMEGA_EL  = 471.23889803846896f;   // 75 * 2*pi
    constexpr float TAU       = 1.0e-4f;
    constexpr float DET_MIN   = 1.0e-8f;
    constexpr float I_CLIP    = 500.0f;

    // ---- shared weight tables (per block) + per-wave activation tiles
    __shared__ _Float16 w2tb[2][4096];          // W2^T in per-lane-contiguous B-frag layout
    __shared__ float    W1s[2][2 * HD];
    __shared__ float    b1s[2][HD];
    __shared__ float    b2s[2][HD];
    __shared__ float    W3s[2][HD];
    __shared__ float    b3s[2];
    __shared__ _Float16 h1t[WAVES_PER_BLK][16 * TSTR];   // h1 (fwd A) / gz2 (bwd A)
    __shared__ _Float16 sp1t[WAVES_PER_BLK][16 * TSTR];  // swish'(z1) / gz1 (in place)
    __shared__ unsigned hs2t[WAVES_PER_BLK][16 * TSTR];  // packed {h2, swish'(z2)}

    const int tid  = threadIdx.x;
    const int lane = tid & 31;
    const int wid  = tid >> 5;

    const float* W2g[2] = {Wd2, Wq2};
    const float* W1g[2] = {Wd1, Wq1};
    const float* b1g[2] = {bd1, bq1};
    const float* b2g[2] = {bd2, bq2};
    const float* W3g[2] = {Wd3, Wq3};

    // ---- block-cooperative setup: W2^T -> WMMA B-operand fragment layout.
    // frag(c,tt): lane ln holds col N = 16*tt + (ln&15), K = c*32 + (ln>>4)*8 + h + (h>>3)*8,
    // stored so each lane's 16 halves are contiguous (2x ds_load_b128 per frag).
    for (int net = 0; net < 2; ++net) {
        for (int idx = tid; idx < 4096; idx += 128) {
            int h  = idx & 15;
            int ln = (idx >> 4) & 31;
            int tt = (idx >> 9) & 3;
            int c  = idx >> 11;
            int K  = c * 32 + (ln >> 4) * 8 + h + (h >> 3) * 8;
            int N  = tt * 16 + (ln & 15);
            w2tb[net][idx] = (_Float16)W2g[net][N * HD + K];   // W2^T[K][N] = W2[N][K]
        }
        for (int idx = tid; idx < 2 * HD; idx += 128) W1s[net][idx] = W1g[net][idx];
        for (int idx = tid; idx < HD; idx += 128) {
            b1s[net][idx] = b1g[net][idx];
            b2s[net][idx] = b2g[net][idx];
            W3s[net][idx] = W3g[net][idx];
        }
    }
    if (tid == 0) { b3s[0] = bd3[0]; b3s[1] = bq3[0]; }
    __syncthreads();

    // ---- register-resident forward W2 B-fragments (128 VGPRs, live all T steps)
    v16h bf[2][2][4];
#pragma unroll
    for (int net = 0; net < 2; ++net)
#pragma unroll
      for (int c = 0; c < 2; ++c)
#pragma unroll
        for (int tt = 0; tt < 4; ++tt) {
            const int kb = c * 32 + (lane >> 4) * 8;
            const int n  = tt * 16 + (lane & 15);
#pragma unroll
            for (int h = 0; h < 16; ++h)
                bf[net][c][tt][h] = (_Float16)W2g[net][(kb + h + (h >> 3) * 8) * HD + n];
        }

    // ---- per-trajectory state (rows keyed by lane&15; both wave halves mirror it)
    const int row = lane & 15;
    const int hf  = lane >> 4;
    const int b   = blockIdx.x * (WAVES_PER_BLK * ROWS_PER_WAVE) + wid * ROWS_PER_WAVE + row;

    float id_amp = init_obs[b * 4 + 0] * I_LIM;
    float iq_amp = init_obs[b * 4 + 1] * I_LIM;
    const float o2 = init_obs[b * 4 + 2];
    const float o3 = init_obs[b * 4 + 3];

    if (lane < 16) {
        float4 r0 = make_float4(id_amp * INV_I_LIM, iq_amp * INV_I_LIM, o2, o3);
        *(float4*)(out + (size_t)b * 4) = r0;
    }

    _Float16* ht = h1t[wid];
    _Float16* st = sp1t[wid];
    unsigned* qt = hs2t[wid];
    const int j0 = hf * 32;
    const float* act_row = actions + (size_t)b * T_STEPS * 2;

    for (int t = 0; t < T_STEPS; ++t) {
        const float xn0 = id_amp * INV_I_LIM;
        const float xn1 = iq_amp * INV_I_LIM;
        const float u0  = act_row[t * 2 + 0] * U_MAX;
        const float u1  = act_row[t * 2 + 1] * U_MAX;
        if (t + 1 < T_STEPS)
            __builtin_prefetch(act_row + (t + 1) * 2, 0, 1);   // global_prefetch_b8

        float ps[2], ga[2], gb[2];

#pragma unroll
        for (int net = 0; net < 2; ++net) {
            // ---- layer 1 forward: h1 = swish(xn @ W1 + b1), keep swish' for VJP
#pragma unroll 8
            for (int jj = 0; jj < 32; ++jj) {
                const int j = j0 + jj;
                float z1 = xn0 * W1s[net][j] + xn1 * W1s[net][HD + j] + b1s[net][j];
                float s  = fast_sig(z1);
                ht[row * TSTR + j] = (_Float16)(z1 * s);
                st[row * TSTR + j] = (_Float16)(s * (1.0f + z1 * (1.0f - s)));
            }

            // ---- layer 2 forward: z2 = h1 @ W2 + b2  (8x v_wmma_f32_16x16x32_f16)
            v16h a0, a1;
            {
                const int kb0 = hf * 8;
#pragma unroll
                for (int h = 0; h < 8; ++h) {
                    a0[h]     = ht[row * TSTR + kb0 + h];
                    a0[h + 8] = ht[row * TSTR + kb0 + 16 + h];
                    a1[h]     = ht[row * TSTR + 32 + kb0 + h];
                    a1[h + 8] = ht[row * TSTR + 32 + kb0 + 16 + h];
                }
            }
            v8f acc[4];
#pragma unroll
            for (int tt = 0; tt < 4; ++tt) {
                const float bb = b2s[net][tt * 16 + row];
                v8f c0;
#pragma unroll
                for (int v = 0; v < 8; ++v) c0[v] = bb;   // bias folded into C
                c0 = __builtin_amdgcn_wmma_f32_16x16x32_f16(false, a0, false, bf[net][0][tt],
                                                            (short)0, c0, false, false);
                acc[tt] = __builtin_amdgcn_wmma_f32_16x16x32_f16(false, a1, false, bf[net][1][tt],
                                                                 (short)0, c0, false, false);
            }

            // ---- swish(z2): pack {h2, swish'(z2)} into one u32/elem (C layout -> tile)
#pragma unroll
            for (int tt = 0; tt < 4; ++tt)
#pragma unroll
              for (int v = 0; v < 8; ++v) {
                const float z2 = acc[tt][v];
                const float s  = fast_sig(z2);
                const _Float16 h2v  = (_Float16)(z2 * s);
                const _Float16 sp2v = (_Float16)(s * (1.0f + z2 * (1.0f - s)));
                const int r2  = hf * 8 + v;
                const int col = tt * 16 + row;
                const unsigned pk = (unsigned)__builtin_bit_cast(unsigned short, h2v)
                                  | ((unsigned)__builtin_bit_cast(unsigned short, sp2v) << 16);
                qt[r2 * TSTR + col] = pk;
              }

            // ---- layer 3 forward: p = tanh(h2 @ W3 + b3); gp = 1 - p^2
            float part = 0.0f;
#pragma unroll 8
            for (int jj = 0; jj < 32; ++jj) {
                const int j = j0 + jj;
                const unsigned pk = qt[row * TSTR + j];
                part += (float)__builtin_bit_cast(_Float16, (unsigned short)(pk & 0xffffu))
                        * W3s[net][j];
            }
            const float z3 = part + __shfl_xor(part, 16, 32) + b3s[net];
            const float p  = fast_tanh(z3);
            const float gp = 1.0f - p * p;

            // ---- gz2 = gp * W3 * swish'(z2)  (A operand for bwd GEMM, reuses h1 tile)
#pragma unroll 8
            for (int jj = 0; jj < 32; ++jj) {
                const int j = j0 + jj;
                const unsigned pk = qt[row * TSTR + j];
                const float sp2 = (float)__builtin_bit_cast(_Float16, (unsigned short)(pk >> 16));
                ht[row * TSTR + j] = (_Float16)(gp * W3s[net][j] * sp2);
            }

            // ---- backward GEMM: gh1 = gz2 @ W2^T  (8x WMMA, B frags from LDS)
            v16h g0, g1;
            {
                const int kb0 = hf * 8;
#pragma unroll
                for (int h = 0; h < 8; ++h) {
                    g0[h]     = ht[row * TSTR + kb0 + h];
                    g0[h + 8] = ht[row * TSTR + kb0 + 16 + h];
                    g1[h]     = ht[row * TSTR + 32 + kb0 + h];
                    g1[h + 8] = ht[row * TSTR + 32 + kb0 + 16 + h];
                }
            }
            v8f gacc[4];
#pragma unroll
            for (int tt = 0; tt < 4; ++tt) {
                v16h bT0, bT1;
#pragma unroll
                for (int h = 0; h < 16; ++h) {
                    bT0[h] = w2tb[net][((0 * 4 + tt) * 32 + lane) * 16 + h];
                    bT1[h] = w2tb[net][((1 * 4 + tt) * 32 + lane) * 16 + h];
                }
                v8f z = {};
                z = __builtin_amdgcn_wmma_f32_16x16x32_f16(false, g0, false, bT0,
                                                           (short)0, z, false, false);
                gacc[tt] = __builtin_amdgcn_wmma_f32_16x16x32_f16(false, g1, false, bT1,
                                                                  (short)0, z, false, false);
            }

            // ---- gz1 = gh1 * swish'(z1) (in place in sp1 tile)
#pragma unroll
            for (int tt = 0; tt < 4; ++tt)
#pragma unroll
              for (int v = 0; v < 8; ++v) {
                const int r2  = hf * 8 + v;
                const int col = tt * 16 + row;
                const float sp1 = (float)st[r2 * TSTR + col];
                st[r2 * TSTR + col] = (_Float16)(gacc[tt][v] * sp1);
              }

            // ---- layer 1 backward: gx_i = (1/I_LIM) * sum_k gz1[k] * W1[i][k]
            float p0 = 0.0f, p1 = 0.0f;
#pragma unroll 8
            for (int jj = 0; jj < 32; ++jj) {
                const int j = j0 + jj;
                const float g = (float)st[row * TSTR + j];
                p0 += g * W1s[net][j];
                p1 += g * W1s[net][HD + j];
            }
            ps[net] = p;
            ga[net] = (p0 + __shfl_xor(p0, 16, 32)) * INV_I_LIM;
            gb[net] = (p1 + __shfl_xor(p1, 16, 32)) * INV_I_LIM;
        }

        // ---- 2x2 flux-Jacobian solve + explicit Euler + clip (mirrored in both halves)
        const float a = ga[0], bgrad = gb[0], c = ga[1], d = gb[1];
        const float pd = ps[0], pq = ps[1];
        float det = a * d - bgrad * c;
        if (fabsf(det) < DET_MIN) det = (det >= 0.0f) ? DET_MIN : -DET_MIN;
        const float rhs0 = u0 - R_S * id_amp + OMEGA_EL * pq;
        const float rhs1 = u1 - R_S * iq_amp - OMEGA_EL * pd;
        const float inv_det = fast_rcp(det);
        const float did = (d * rhs0 - bgrad * rhs1) * inv_det;
        const float diq = (a * rhs1 - c * rhs0) * inv_det;
        id_amp = fminf(fmaxf(id_amp + TAU * did, -I_CLIP), I_CLIP);
        iq_amp = fminf(fmaxf(iq_amp + TAU * diq, -I_CLIP), I_CLIP);

        if (lane < 16) {
            float4 r = make_float4(id_amp * INV_I_LIM, iq_amp * INV_I_LIM, o2, o3);
            *(float4*)(out + ((size_t)(t + 1) * B_TOT + b) * 4) = r;
        }
    }
}

extern "C" void kernel_launch(void* const* d_in, const int* in_sizes, int n_in,
                              void* d_out, int out_size, void* d_ws, size_t ws_size,
                              hipStream_t stream) {
    (void)in_sizes; (void)n_in; (void)out_size; (void)d_ws; (void)ws_size;
    const float* init_obs = (const float*)d_in[0];
    const float* actions  = (const float*)d_in[1];
    const float* Wd1 = (const float*)d_in[2];
    const float* bd1 = (const float*)d_in[3];
    const float* Wd2 = (const float*)d_in[4];
    const float* bd2 = (const float*)d_in[5];
    const float* Wd3 = (const float*)d_in[6];
    const float* bd3 = (const float*)d_in[7];
    const float* Wq1 = (const float*)d_in[8];
    const float* bq1 = (const float*)d_in[9];
    const float* Wq2 = (const float*)d_in[10];
    const float* bq2 = (const float*)d_in[11];
    const float* Wq3 = (const float*)d_in[12];
    const float* bq3 = (const float*)d_in[13];
    float* out = (float*)d_out;

    dim3 grid(B_TOT / (WAVES_PER_BLK * ROWS_PER_WAVE));   // 256 blocks
    dim3 block(WAVES_PER_BLK * 32);                       // 128 threads = 4 wave32
    motor_rollout_kernel<<<grid, block, 0, stream>>>(
        init_obs, actions,
        Wd1, bd1, Wd2, bd2, Wd3, bd3,
        Wq1, bq1, Wq2, bq2, Wq3, bq3,
        out);
}
